// VanillaRNN_47339129536586
// MI455X (gfx1250) — matrix operand order repaired and checked by reading, not compiled, not run on hardware
//
#include <hip/hip_runtime.h>
#include <hip/hip_bf16.h>

// ---------------------------------------------------------------------------
// VanillaRNN on gfx1250 (MI455X):
//   h_t = tanh(proj[x_t] + h_{t-1} @ W_hh^T),  proj[v] = emb[v] @ W_ih^T
//   out = last_h @ W_fc^T + b_fc
// 16-batch-row tile per workgroup, 8 wave32 x 16-col slices,
// v_wmma_f32_16x16x32_f16 over K=128 (4 chunks), LDS double-buffered h.
// Round-3 changes: WMMA chain starts from C=0 and the proj gather is folded in
// AFTER the chain (decouples the token->gather chain from WMMA0), plus
// sched_group_barrier hints to issue all DS reads before the WMMA chain so
// waitcnts become partial and WMMAs overlap outstanding loads.
// ---------------------------------------------------------------------------

#define VOCAB   60
#define VOCABP  64          // padded vocab rows
#define HIDDEN  128
#define BATCH   1024
#define SEQLEN  256
#define TILE_M  16          // batch rows per workgroup

typedef __attribute__((ext_vector_type(16))) _Float16 v16h;
typedef __attribute__((ext_vector_type(8)))  _Float16 v8h;
typedef __attribute__((ext_vector_type(8)))  float    v8f;

__device__ __forceinline__ float fast_tanh(float x) {
#if __has_builtin(__builtin_amdgcn_tanhf)
  return __builtin_amdgcn_tanhf(x);
#elif __has_builtin(__builtin_amdgcn_tanh_f32)
  return __builtin_amdgcn_tanh_f32(x);
#else
  return tanhf(x);
#endif
}

// Scheduling hint for the recurrence step body:
//   group 1: all 18 DS reads (2 token b128 + 8 A b128 + 8 proj b32)
//   group 2: the 4 WMMAs
//   group 3: the 8 DS writes (h publish)
__device__ __forceinline__ void sched_hint_step() {
#if __has_builtin(__builtin_amdgcn_sched_group_barrier)
  __builtin_amdgcn_sched_group_barrier(0x100, 18, 0);  // DS_READ
  __builtin_amdgcn_sched_group_barrier(0x008,  4, 0);  // MFMA/WMMA
  __builtin_amdgcn_sched_group_barrier(0x200,  8, 0);  // DS_WRITE
#endif
}

// ---------------------------------------------------------------------------
// Precompute: proj[v][j] = dot(emb[v], W_ih[j])   (v padded to 64, pad rows 0)
//             whh16      = f16(W_hh)              (128x128, row-major)
//             wfc16      = f16(W_fc padded to 64 rows)
// ---------------------------------------------------------------------------
__global__ __launch_bounds__(256) void rnn_precompute(
    const float* __restrict__ emb, const float* __restrict__ W_ih,
    const float* __restrict__ W_hh, const float* __restrict__ W_fc,
    float* __restrict__ proj, _Float16* __restrict__ whh16,
    _Float16* __restrict__ wfc16) {
  const int id = blockIdx.x * blockDim.x + threadIdx.x;   // 0 .. 16383

  if (id < HIDDEN * HIDDEN)                               // 16384 converts
    whh16[id] = (_Float16)W_hh[id];

  if (id < VOCABP * HIDDEN) {                             // 8192 dots + converts
    const int v = id >> 7;
    const int j = id & (HIDDEN - 1);
    float s = 0.0f;
    if (v < VOCAB) {
      const float* ev = emb + v * HIDDEN;
      const float* wj = W_ih + j * HIDDEN;
      #pragma unroll 8
      for (int k = 0; k < HIDDEN; ++k) s = fmaf(ev[k], wj[k], s);
    }
    proj[id]  = s;
    wfc16[id] = (v < VOCAB) ? (_Float16)W_fc[id] : (_Float16)0.0f;
  }
}

// ---------------------------------------------------------------------------
// Main recurrence kernel: 64 blocks x 256 threads (8 wave32).
// ---------------------------------------------------------------------------
__global__ __launch_bounds__(256) void rnn_main(
    const int* __restrict__ x, const int* __restrict__ lengths,
    const float* __restrict__ proj, const _Float16* __restrict__ whh16,
    const _Float16* __restrict__ wfc16, const float* __restrict__ b_fc,
    float* __restrict__ out) {
  __shared__ __align__(16) float    s_proj[VOCABP * HIDDEN];      // 32 KB
  __shared__ __align__(16) int      s_xt[SEQLEN * TILE_M];        // 16 KB, time-major
  __shared__ __align__(32) _Float16 s_h[2][TILE_M * HIDDEN];      //  8 KB
  __shared__ __align__(32) _Float16 s_last[TILE_M * HIDDEN];      //  4 KB
  __shared__ int s_len[TILE_M];
  __shared__ int s_tmax;

  const int tid     = threadIdx.x;
  const int wave    = tid >> 5;
  const int lane    = tid & 31;
  const int rowbase = blockIdx.x * TILE_M;

  // Stage proj table (8192 f32) into LDS.
  {
    const float4* src = (const float4*)proj;
    float4*       dst = (float4*)s_proj;
    #pragma unroll 2
    for (int i = tid; i < (VOCABP * HIDDEN) / 4; i += 256) dst[i] = src[i];
  }
  // Stage token tile TRANSPOSED to time-major: s_xt[t*16 + row].
  // Global reads stay coalesced (consecutive tid -> consecutive t).
  for (int i = tid; i < TILE_M * SEQLEN; i += 256) {
    const int row = i >> 8;           // 0..15
    const int t   = i & (SEQLEN - 1); // 0..255
    s_xt[t * TILE_M + row] = x[(rowbase + row) * SEQLEN + t];
  }
  if (tid < TILE_M) s_len[tid] = lengths[rowbase + tid];
  for (int i = tid; i < TILE_M * HIDDEN; i += 256) s_h[0][i] = (_Float16)0.0f;
  __syncthreads();

  if (tid == 0) {
    int mx = 1;
    for (int r = 0; r < TILE_M; ++r) mx = (s_len[r] > mx) ? s_len[r] : mx;
    s_tmax = mx;
  }

  // Per-lane geometry (ISA §7.12.2 row/col layouts; K split contiguous, k = 16*(lane/16) + s).
  const int col  = (wave << 4) + (lane & 15);   // D/N column owned by this lane
  const int m    = lane & 15;                   // A row read by this lane
  const int mhi  = (lane < 16) ? 0 : 8;         // C/D row offset
  const int aoff = (lane < 16) ? 0 : 16;        // A K sub-offset
  const int boff = (lane < 16) ? 0 : 16;        // B K sub-offset

  // W_hh^T as WMMA B operand: lane holds column j=col, 16 contiguous K per chunk.
  v16h bw[4];
  {
    const _Float16* wrow = whh16 + col * HIDDEN;
    #pragma unroll
    for (int c = 0; c < 4; ++c)
      bw[c] = *(const v16h*)(wrow + 32 * c + boff);
  }
  __syncthreads();

  int lent1[8];                       // len-1 per owned row
  #pragma unroll
  for (int r = 0; r < 8; ++r) lent1[r] = s_len[r + mhi] - 1;
  const int tmax = s_tmax;

  _Float16 lastv[8];                  // register-resident last-h capture
  #pragma unroll
  for (int r = 0; r < 8; ++r) lastv[r] = (_Float16)0.0f;

  // ---------------- serial recurrence ----------------
  for (int t = 0; t < tmax; ++t) {
    const int rbuf = t & 1;
    const int wbuf = rbuf ^ 1;

    // Chain 1: token gather (2x ds_load_b128) -> proj gather (8x ds_load_b32).
    const int4 tk0 = *(const int4*)&s_xt[t * TILE_M + mhi];
    const int4 tk1 = *(const int4*)&s_xt[t * TILE_M + mhi + 4];
    const int tok[8] = {tk0.x, tk0.y, tk0.z, tk0.w,
                        tk1.x, tk1.y, tk1.z, tk1.w};

    // Chain 2 (independent): A chunks (8x ds_load_b128).
    v16h a[4];
    #pragma unroll
    for (int c = 0; c < 4; ++c) {
      const _Float16* hrow = &s_h[rbuf][m * HIDDEN + 32 * c + aoff];
      const v8h lo = *(const v8h*)(hrow);
      const v8h hi = *(const v8h*)(hrow + 8);
      #pragma unroll
      for (int i = 0; i < 8; ++i) { a[c][i] = lo[i]; a[c][i + 8] = hi[i]; }
    }

    float pv[8];
    #pragma unroll
    for (int r = 0; r < 8; ++r)
      pv[r] = s_proj[tok[r] * HIDDEN + col];

    // h_{t-1} @ W_hh^T starting from C=0: WMMA0 depends only on A chunk 0,
    // so the chain overlaps the in-flight proj gathers.
    v8f acc;
    #pragma unroll
    for (int i = 0; i < 8; ++i) acc[i] = 0.0f;
    #pragma unroll
    for (int c = 0; c < 4; ++c)
      acc = __builtin_amdgcn_wmma_f32_16x16x32_f16(
          false, a[c], false, bw[c], (short)0, acc, false, false);

    // z = (h W_hh^T) + proj[x_t]; tanh; publish; capture last-valid h.
    #pragma unroll
    for (int r = 0; r < 8; ++r) {
      const float    h   = fast_tanh(acc[r] + pv[r]);
      const _Float16 h16 = (_Float16)h;
      s_h[wbuf][(r + mhi) * HIDDEN + col] = h16;
      lastv[r] = (lent1[r] == t) ? h16 : lastv[r];
    }
    sched_hint_step();
    __syncthreads();
  }

  // Publish captured last-h once for the FC exchange.
  #pragma unroll
  for (int r = 0; r < 8; ++r)
    s_last[(r + mhi) * HIDDEN + col] = lastv[r];
  __syncthreads();

  // ---------------- final FC: last_h @ W_fc^T + b_fc ----------------
  if (wave < 4) {                                // cols 0..63 (valid: <60)
    v16h bf[4];
    {
      const _Float16* wrow = wfc16 + col * HIDDEN;
      #pragma unroll
      for (int c = 0; c < 4; ++c)
        bf[c] = *(const v16h*)(wrow + 32 * c + boff);
    }
    v8f acc;
    #pragma unroll
    for (int i = 0; i < 8; ++i) acc[i] = 0.0f;
    #pragma unroll
    for (int c = 0; c < 4; ++c) {
      const _Float16* hrow = &s_last[m * HIDDEN + 32 * c + aoff];
      const v8h lo = *(const v8h*)(hrow);
      const v8h hi = *(const v8h*)(hrow + 8);
      v16h a;
      #pragma unroll
      for (int i = 0; i < 8; ++i) { a[i] = lo[i]; a[i + 8] = hi[i]; }
      acc = __builtin_amdgcn_wmma_f32_16x16x32_f16(
          false, a, false, bf[c], (short)0, acc, false, false);
    }
    if (col < VOCAB) {
      const float bias = b_fc[col];
      #pragma unroll
      for (int r = 0; r < 8; ++r)
        out[(rowbase + r + mhi) * VOCAB + col] = acc[r] + bias;
    }
  }
}

// ---------------------------------------------------------------------------
// Host-side launch. Inputs (setup_inputs order):
//   0:x(i32 [1024,256]) 1:lengths(i32 [1024]) 2:emb(f32 [60,128])
//   3:W_ih(f32 [128,128]) 4:W_hh(f32 [128,128]) 5:W_fc(f32 [60,128]) 6:b_fc(f32 [60])
// Output: f32 [1024,60].
// ---------------------------------------------------------------------------
extern "C" void kernel_launch(void* const* d_in, const int* in_sizes, int n_in,
                              void* d_out, int out_size, void* d_ws, size_t ws_size,
                              hipStream_t stream) {
  const int*   x       = (const int*)d_in[0];
  const int*   lengths = (const int*)d_in[1];
  const float* emb     = (const float*)d_in[2];
  const float* W_ih    = (const float*)d_in[3];
  const float* W_hh    = (const float*)d_in[4];
  const float* W_fc    = (const float*)d_in[5];
  const float* b_fc    = (const float*)d_in[6];
  float*       out     = (float*)d_out;

  // Workspace carving (hipMalloc base is 256B aligned):
  //   [0,            32768)  proj   f32 [64,128]
  //   [32768,        65536)  whh16  f16 [128,128]
  //   [65536,        81920)  wfc16  f16 [64,128]
  char* ws = (char*)d_ws;
  float*    proj  = (float*)(ws);
  _Float16* whh16 = (_Float16*)(ws + VOCABP * HIDDEN * sizeof(float));
  _Float16* wfc16 = (_Float16*)(ws + VOCABP * HIDDEN * sizeof(float)
                                   + HIDDEN * HIDDEN * sizeof(_Float16));

  rnn_precompute<<<64, 256, 0, stream>>>(emb, W_ih, W_hh, W_fc,
                                         proj, whh16, wfc16);
  rnn_main<<<BATCH / TILE_M, 256, 0, stream>>>(x, lengths, proj, whh16,
                                               wfc16, b_fc, out);
}
